// MHSelfAttention_with_ln_res_69295002353948
// MI455X (gfx1250) — matrix operand order repaired and checked by reading, not compiled, not run on hardware
//
#include <hip/hip_runtime.h>
#include <hip/hip_bf16.h>

typedef __attribute__((ext_vector_type(16))) _Float16 v16h;
typedef __attribute__((ext_vector_type(8)))  float    v8f;

#define DEV static __device__ __forceinline__

constexpr int DDIM  = 512;
constexpr int NSEQ  = 325;
constexpr int NP    = 352;              // 325 padded to 11*32
constexpr int BT    = 96;               // B*T
constexpr int NHEAD = 8;
constexpr int DK    = 64;
constexpr int BTH   = BT * NHEAD;       // 768
constexpr int TOK   = BT * NSEQ;        // 31200

// ---- workspace layout (bytes) ----
constexpr size_t SZ_HEADBUF = (size_t)BTH * NP * DK * 2;   // 34,603,008 each
constexpr size_t OFF_Q   = 0;                               // f16 [bth][n<NP][dk]
constexpr size_t OFF_KT  = OFF_Q  + SZ_HEADBUF;             // f16 kT, fragment-ready
constexpr size_t OFF_V   = OFF_KT + SZ_HEADBUF;             // f16 v,  fragment-ready
constexpr size_t OFF_O   = OFF_V  + SZ_HEADBUF;             // f16 [token][512]
constexpr size_t OFF_PRE = 0;                               // f32 [token][512], reuses q/kT

constexpr int FRAG_DW = 256;            // dwords per fragment-ready 32x16 B block

union Frag  { v16h v; unsigned u[8]; };
union Pack4 { _Float16 h[4]; uint2 u; };
union Pack2 { _Float16 h[2]; unsigned u; };

// A-matrix 16x32 f16 fragment from a row-major f16 buffer (viewed as dwords).
// Layout: lanes 0-15 -> K {0..15}; lanes 16-31 -> K {16..31} (matches B fragments).
DEV v16h load_a_frag(const unsigned* base, int stride_dw, int lane) {
  const int m    = lane & 15;
  const int ksel = (lane >> 4) & 1;
  const unsigned* p = base + (size_t)m * stride_dw + ksel * 8;
  Frag f;
#pragma unroll
  for (int i = 0; i < 8; ++i) f.u[i] = p[i];
  return f.v;
}

// B-matrix 32x16 f16 fragment from a fragment-ready block: lane's 8 dwords
// contiguous -> lowers to two b128 loads, no register gathering.
// dword [lane*8 + i] = halves (k = (lane>>4)*16 + 2i, +1) at column n = lane&15.
DEV v16h load_b_frag_fr(const unsigned* base, int lane) {
  const unsigned* p = base + lane * 8;
  Frag f;
#pragma unroll
  for (int i = 0; i < 8; ++i) f.u[i] = p[i];
  return f.v;
}

// scatter-store address helper for fragment-ready blocks:
// element (k_local in [0,32), n_local in [0,16)) of one block
DEV size_t fr_half_idx(size_t block, int k_local, int n_local) {
  int lane = n_local + 16 * (k_local >> 4);
  int ii   = (k_local & 15) >> 1;
  return (block * 32 + lane) * 16 + ii * 2 + (k_local & 1);
}

DEV v8f wmma_f16(v16h a, v16h b, v8f c) {
  return __builtin_amdgcn_wmma_f32_16x16x32_f16(false, a, false, b, (short)0, c, false, false);
}

// ---------------------------------------------------------------- K0: zero pad buffers
__global__ void __launch_bounds__(256) k0_zero(uint4* __restrict__ p, long n) {
  long i = (long)blockIdx.x * blockDim.x + threadIdx.x;
  if (i < n) p[i] = make_uint4(0u, 0u, 0u, 0u);
}

// ---------------------------------------------------------------- K1: QKV projection
// out = In @ W_head[mat] + bias[mat]; scatter into head-split q / kT / v buffers.
// Block: 256 thr, 8 waves; each wave: 32 rows x 64 cols; block tile 256x64.
__global__ void __launch_bounds__(256) k1_proj(
    const float* __restrict__ Qin, const float* __restrict__ Kin,
    const float* __restrict__ Vin, const float* __restrict__ Wh,
    const float* __restrict__ bias, char* __restrict__ ws)
{
  __shared__ __align__(16) _Float16 la[256 * 32];   // A tile, row-major, 16 KB
  __shared__ __align__(16) _Float16 lb[32 * 64];    // B tile, fragment-ready, 4 KB

  const int tid  = threadIdx.x;
  const int wave = tid >> 5, lane = tid & 31;
  const int row0 = blockIdx.x * 256;
  const int col0 = blockIdx.y * 64;
  const int mat  = blockIdx.z;
  const float* In = (mat == 0) ? Qin : (mat == 1) ? Kin : Vin;
  const float* W  = Wh + (size_t)mat * DDIM * DDIM;
  const float* bv = bias + mat * DDIM;

  v8f acc[2][4] = {{v8f{}, v8f{}, v8f{}, v8f{}}, {v8f{}, v8f{}, v8f{}, v8f{}}};

  for (int kk = 0; kk < DDIM; kk += 32) {
    // stage A: 256x32 f32 -> f16, float4 loads, packed uint2 LDS stores
#pragma unroll
    for (int it = 0; it < 8; ++it) {
      int i  = tid + it * 256;          // float4 index, 2048 total
      int r  = i >> 3, c4 = i & 7;      // 8 float4 per 32-wide row
      int gr = row0 + r;
      gr = (gr < TOK) ? gr : (TOK - 1); // clamp, no divergent branch
      float4 f = ((const float4*)In)[(size_t)gr * (DDIM / 4) + (kk >> 2) + c4];
      Pack4 pk;
      pk.h[0] = (_Float16)f.x; pk.h[1] = (_Float16)f.y;
      pk.h[2] = (_Float16)f.z; pk.h[3] = (_Float16)f.w;
      ((uint2*)la)[r * 8 + c4] = pk.u;
    }
    // stage B: 32x64; load both k's of a pair, one packed b32 store (fragment-ready)
#pragma unroll
    for (int it = 0; it < 4; ++it) {
      int i = tid + it * 256;           // pair index*64 + n, 1024 total
      int p = i >> 6, n = i & 63;
      float w0 = W[(size_t)(kk + 2 * p)     * DDIM + col0 + n];
      float w1 = W[(size_t)(kk + 2 * p + 1) * DDIM + col0 + n];
      Pack2 pk; pk.h[0] = (_Float16)w0; pk.h[1] = (_Float16)w1;
      int nt = n >> 4, ln = (n & 15) + 16 * (p >> 3), ii = p & 7;
      ((unsigned*)lb)[nt * FRAG_DW + ln * 8 + ii] = pk.u;
    }
    // prefetch next K tile (speculative, uniform guard)
    if (kk + 32 < DDIM) {
      int r0 = row0 + (tid >> 3); r0 = (r0 < TOK) ? r0 : (TOK - 1);
      __builtin_prefetch(&((const float4*)In)[(size_t)r0 * (DDIM / 4) + ((kk + 32) >> 2) + (tid & 7)], 0, 0);
      __builtin_prefetch(&W[(size_t)(kk + 32 + (tid >> 6)) * DDIM + col0 + (tid & 63)], 0, 0);
    }
    __syncthreads();
    const unsigned* au = (const unsigned*)la + wave * 32 * 16;
    v16h a0 = load_a_frag(au,           16, lane);
    v16h a1 = load_a_frag(au + 16 * 16, 16, lane);
    const unsigned* bu = (const unsigned*)lb;
#pragma unroll
    for (int t = 0; t < 4; ++t) {
      v16h b = load_b_frag_fr(bu + t * FRAG_DW, lane);
      acc[0][t] = wmma_f16(a0, b, acc[0][t]);
      acc[1][t] = wmma_f16(a1, b, acc[1][t]);
    }
    __syncthreads();
  }

  _Float16* qb = (_Float16*)(ws + OFF_Q);
  _Float16* kb = (_Float16*)(ws + OFF_KT);
  _Float16* vb = (_Float16*)(ws + OFF_V);
  const int hi = lane >> 4, lo = lane & 15;
#pragma unroll
  for (int sb = 0; sb < 2; ++sb) {
    const int rbase = row0 + wave * 32 + sb * 16 + hi * 8;
    int btr[8], nr[8];
#pragma unroll
    for (int r = 0; r < 8; ++r) {
      int token = rbase + r;
      btr[r] = (token < TOK) ? token / NSEQ : -1;
      nr[r]  = (token < TOK) ? token % NSEQ : 0;
    }
#pragma unroll
    for (int t = 0; t < 4; ++t) {
      int e = col0 + t * 16 + lo;
      float bvv = bv[e];
      int h = e & 7, j = e >> 3;                    // col e = j*8 + h (head interleave)
#pragma unroll
      for (int r = 0; r < 8; ++r) {
        if (btr[r] < 0) continue;
        float val = acc[sb][t][r] + bvv;
        size_t bth = (size_t)btr[r] * NHEAD + h;
        int n = nr[r];
        if (mat == 0) {
          qb[(bth * NP + n) * DK + j] = (_Float16)(val * 0.125f);        // fold 1/sqrt(dk)
        } else if (mat == 1) {
          // kT fragment-ready: blocks [bth][ks=j>>5][nt=n>>4], K-local = j&31, col = n&15
          size_t block = (bth * 2 + (j >> 5)) * 22 + (n >> 4);
          kb[fr_half_idx(block, j & 31, n & 15)] = (_Float16)val;
        } else {
          // v fragment-ready: blocks [bth][ks=n>>5][dt=j>>4], K-local = n&31, col = j&15
          size_t block = (bth * 11 + (n >> 5)) * 4 + (j >> 4);
          vb[fr_half_idx(block, n & 31, j & 15)] = (_Float16)val;
        }
      }
    }
  }
}

// ---------------------------------------------------------------- K2: attention per (b,t,h)
__global__ void __launch_bounds__(128) k2_attn(char* __restrict__ ws)
{
  __shared__ __align__(16) _Float16 lp[4][16 * NP];   // per-wave P slab, 45 KB
  const int tid  = threadIdx.x;
  const int wave = tid >> 5, lane = tid & 31;
  const int bth  = blockIdx.x;
  const int qt   = blockIdx.y * 4 + wave;
  if (qt >= NP / 16) return;            // 22 q-tiles; no cross-wave LDS sharing -> safe

  const int lo = lane & 15, hi = lane >> 4;
  const unsigned* qu  = (const unsigned*)(ws + OFF_Q)  + ((size_t)bth * NP + qt * 16) * (DK / 2);
  const unsigned* ktu = (const unsigned*)(ws + OFF_KT) + (size_t)bth * (2 * 22 * FRAG_DW);
  const unsigned* vu  = (const unsigned*)(ws + OFF_V)  + (size_t)bth * (11 * 4 * FRAG_DW);

  v16h a0 = load_a_frag(qu,      DK / 2, lane);       // K = 0..31 of d_k
  v16h a1 = load_a_frag(qu + 16, DK / 2, lane);       // K = 32..63

  v8f S[22];
#pragma unroll
  for (int nt = 0; nt < 22; ++nt) {                   // S = q . kT  (44 WMMAs)
    v16h b0 = load_b_frag_fr(ktu + (0 * 22 + nt) * FRAG_DW, lane);
    v16h b1 = load_b_frag_fr(ktu + (1 * 22 + nt) * FRAG_DW, lane);
    v8f s = v8f{};
    s = wmma_f16(a0, b0, s);
    s = wmma_f16(a1, b1, s);
    S[nt] = s;
  }

#pragma unroll
  for (int nt = 20; nt < 22; ++nt) {                  // mask padded key columns
    if (nt * 16 + lo >= NSEQ) {
#pragma unroll
      for (int r = 0; r < 8; ++r) S[nt][r] = -3.0e38f;
    }
  }

  float inv[8];
#pragma unroll
  for (int r = 0; r < 8; ++r) {                       // row softmax (16-lane halves)
    float mx = S[0][r];
#pragma unroll
    for (int nt = 1; nt < 22; ++nt) mx = fmaxf(mx, S[nt][r]);
#pragma unroll
    for (int off = 1; off <= 8; off <<= 1) mx = fmaxf(mx, __shfl_xor(mx, off, 32));
    float sum = 0.f;
#pragma unroll
    for (int nt = 0; nt < 22; ++nt) {
      float e = __expf(S[nt][r] - mx);
      S[nt][r] = e;
      sum += e;
    }
#pragma unroll
    for (int off = 1; off <= 8; off <<= 1) sum += __shfl_xor(sum, off, 32);
    inv[r] = 1.0f / sum;
  }

  _Float16* pw = lp[wave];                            // P -> LDS (C-layout -> A-layout)
#pragma unroll
  for (int nt = 0; nt < 22; ++nt) {
#pragma unroll
    for (int r = 0; r < 8; ++r)
      pw[(r + 8 * hi) * NP + nt * 16 + lo] = (_Float16)(S[nt][r] * inv[r]);
  }
  asm volatile("s_wait_dscnt 0" ::: "memory");        // same-wave DS RAW fence

  const unsigned* pu = (const unsigned*)pw;
  v8f O[4] = {v8f{}, v8f{}, v8f{}, v8f{}};
#pragma unroll
  for (int ks = 0; ks < 11; ++ks) {                   // O = P . v  (44 WMMAs)
    v16h ap = load_a_frag(pu + ks * 16, NP / 2, lane);
#pragma unroll
    for (int dt = 0; dt < 4; ++dt) {
      v16h bb = load_b_frag_fr(vu + (ks * 4 + dt) * FRAG_DW, lane);
      O[dt] = wmma_f16(ap, bb, O[dt]);
    }
  }

  _Float16* ob = (_Float16*)(ws + OFF_O);
  const int bt = bth >> 3, h = bth & 7;
#pragma unroll
  for (int dt = 0; dt < 4; ++dt) {
    int j = dt * 16 + lo;
    int e = j * 8 + h;                                // un-interleave heads
#pragma unroll
    for (int r = 0; r < 8; ++r) {
      int n = qt * 16 + r + 8 * hi;
      if (n < NSEQ)
        ob[((size_t)bt * NSEQ + n) * DDIM + e] = (_Float16)O[dt][r];
    }
  }
}

// ---------------------------------------------------------------- K3: out-proj + residual
// Block 256 thr, 8 waves; wave: 32 rows x 64 cols; A direct from f16 o buffer.
__global__ void __launch_bounds__(256) k3_outproj(
    const float* __restrict__ Wout, const float* __restrict__ Xres, char* __restrict__ ws)
{
  __shared__ __align__(16) _Float16 lb[32 * 64];      // fragment-ready
  const int tid = threadIdx.x, wave = tid >> 5, lane = tid & 31;
  const int row0 = blockIdx.x * 256, col0 = blockIdx.y * 64;
  int rb0 = row0 + wave * 32;
  int rc0 = (rb0      <= TOK - 16) ? rb0        : (TOK - 16);
  int rc1 = (rb0 + 16 <= TOK - 16) ? (rb0 + 16) : (TOK - 16);
  const unsigned* ou = (const unsigned*)(ws + OFF_O);
  const unsigned* ab0 = ou + (size_t)rc0 * (DDIM / 2);
  const unsigned* ab1 = ou + (size_t)rc1 * (DDIM / 2);

  v8f acc[2][4] = {{v8f{}, v8f{}, v8f{}, v8f{}}, {v8f{}, v8f{}, v8f{}, v8f{}}};
  for (int kk = 0; kk < DDIM; kk += 32) {
#pragma unroll
    for (int it = 0; it < 4; ++it) {                  // stage W_out tile, fragment-ready
      int i = tid + it * 256;
      int p = i >> 6, n = i & 63;
      float w0 = Wout[(size_t)(kk + 2 * p)     * DDIM + col0 + n];
      float w1 = Wout[(size_t)(kk + 2 * p + 1) * DDIM + col0 + n];
      Pack2 pk; pk.h[0] = (_Float16)w0; pk.h[1] = (_Float16)w1;
      int nt = n >> 4, ln = (n & 15) + 16 * (p >> 3), ii = p & 7;
      ((unsigned*)lb)[nt * FRAG_DW + ln * 8 + ii] = pk.u;
    }
    if (kk + 32 < DDIM)
      __builtin_prefetch(&Wout[(size_t)(kk + 32 + (tid >> 6)) * DDIM + col0 + (tid & 63)], 0, 0);
    __syncthreads();
    v16h a0 = load_a_frag(ab0 + kk / 2, DDIM / 2, lane);
    v16h a1 = load_a_frag(ab1 + kk / 2, DDIM / 2, lane);
    const unsigned* bu = (const unsigned*)lb;
#pragma unroll
    for (int t = 0; t < 4; ++t) {
      v16h b = load_b_frag_fr(bu + t * FRAG_DW, lane);
      acc[0][t] = wmma_f16(a0, b, acc[0][t]);
      acc[1][t] = wmma_f16(a1, b, acc[1][t]);
    }
    __syncthreads();
  }

  float* pre = (float*)(ws + OFF_PRE);
  const int lo = lane & 15, hi = lane >> 4;
#pragma unroll
  for (int sb = 0; sb < 2; ++sb) {
#pragma unroll
    for (int t = 0; t < 4; ++t) {
      int e = col0 + t * 16 + lo;
#pragma unroll
      for (int r = 0; r < 8; ++r) {
        int token = row0 + wave * 32 + sb * 16 + hi * 8 + r;
        if (token < TOK)
          pre[(size_t)token * DDIM + e] = acc[sb][t][r] + Xres[(size_t)token * DDIM + e];
      }
    }
  }
}

// ---------------------------------------------------------------- K4: LayerNorm
__global__ void __launch_bounds__(256) k4_ln(const char* __restrict__ ws,
                                             const float* __restrict__ g,
                                             const float* __restrict__ b,
                                             float* __restrict__ out)
{
  __shared__ float rs[256], rq[256];
  const float* pre = (const float*)(ws + OFF_PRE);
  int row = blockIdx.x, tid = threadIdx.x;
  float a = pre[(size_t)row * DDIM + tid];
  float c = pre[(size_t)row * DDIM + 256 + tid];
  rs[tid] = a + c;
  rq[tid] = a * a + c * c;
  __syncthreads();
  for (int s = 128; s > 0; s >>= 1) {
    if (tid < s) { rs[tid] += rs[tid + s]; rq[tid] += rq[tid + s]; }
    __syncthreads();
  }
  float mu   = rs[0] * (1.0f / DDIM);
  float var  = rq[0] * (1.0f / DDIM) - mu * mu;
  float rstd = rsqrtf(var + 1e-5f);
  out[(size_t)row * DDIM + tid]       = (a - mu) * rstd * g[tid] + b[tid];
  out[(size_t)row * DDIM + 256 + tid] = (c - mu) * rstd * g[tid + 256] + b[tid + 256];
}

// ---------------------------------------------------------------- launch
extern "C" void kernel_launch(void* const* d_in, const int* in_sizes, int n_in,
                              void* d_out, int out_size, void* d_ws, size_t ws_size,
                              hipStream_t stream) {
  const float* X    = (const float*)d_in[0];
  const float* Qi   = (const float*)d_in[1];
  const float* Ki   = (const float*)d_in[2];
  const float* Vi   = (const float*)d_in[3];
  const float* Wh   = (const float*)d_in[4];
  const float* bias = (const float*)d_in[5];
  const float* Wout = (const float*)d_in[6];
  const float* g    = (const float*)d_in[7];
  const float* bta  = (const float*)d_in[8];
  char* ws = (char*)d_ws;

  long nz = (long)(3 * SZ_HEADBUF) / 16;
  k0_zero<<<dim3((unsigned)((nz + 255) / 256)), 256, 0, stream>>>((uint4*)ws, nz);
  k1_proj<<<dim3(122, 8, 3), 256, 0, stream>>>(Qi, Ki, Vi, Wh, bias, ws);
  k2_attn<<<dim3(BTH, 6), 128, 0, stream>>>(ws);
  k3_outproj<<<dim3(122, 8), 256, 0, stream>>>(Wout, X, ws);
  k4_ln<<<TOK, 256, 0, stream>>>(ws, g, bta, (float*)d_out);
}
